// KNN_Embedding_V_58849641890551
// MI455X (gfx1250) — compile-verified, matrix operand order checked by reading, not run on hardware
//
#include <hip/hip_runtime.h>

typedef __attribute__((ext_vector_type(16))) _Float16 v16h;
typedef __attribute__((ext_vector_type(8)))  float    v8f;
typedef __attribute__((ext_vector_type(4)))  float    f32x4;

#define B_      4
#define N_      8192
#define D_      64
#define K_      27
#define E_      256
#define KD_     (K_ * D_)       // 1728
#define NCHUNK  (KD_ / 32)      // 54 chunks of K=32

// ---------------------------------------------------------------------------
// Kernel 1: brute-force KNN (top-27 by squared distance, ascending, self incl)
// One thread per query point; candidates staged through LDS in float4 tiles.
// ---------------------------------------------------------------------------
__global__ __launch_bounds__(256) void knn_kernel(const float* __restrict__ xv,
                                                  int* __restrict__ idx_out) {
  const int batch = blockIdx.x >> 5;                       // 32 blocks / batch
  const int q     = ((blockIdx.x & 31) << 8) + threadIdx.x;
  const float* base = xv + (size_t)batch * N_ * 3;

  const float qx = base[q * 3 + 0];
  const float qy = base[q * 3 + 1];
  const float qz = base[q * 3 + 2];

  float dist[K_];
  int   id[K_];
#pragma unroll
  for (int j = 0; j < K_; ++j) { dist[j] = 3.4e38f; id[j] = 0; }

  __shared__ float4 tile[256];

  for (int t = 0; t < N_ / 256; ++t) {
    __syncthreads();
    {
      const int c = (t << 8) + threadIdx.x;
      tile[threadIdx.x] = make_float4(base[c * 3 + 0], base[c * 3 + 1],
                                      base[c * 3 + 2], 0.0f);
    }
    __syncthreads();

#pragma unroll 4
    for (int i = 0; i < 256; ++i) {
      const float4 p = tile[i];
      const float dx = qx - p.x;
      const float dy = qy - p.y;
      const float dz = qz - p.z;
      const float d  = dx * dx + dy * dy + dz * dz;
      if (d < dist[K_ - 1]) {            // rare after warm-up; EXEC-masked chain
        dist[K_ - 1] = d;
        id[K_ - 1]   = (t << 8) + i;
        // one bubble pass restores full sorted order (list was sorted before)
#pragma unroll
        for (int j = K_ - 1; j > 0; --j) {
          const bool  sw = dist[j] < dist[j - 1];
          const float fd = dist[j - 1];
          const int   fi = id[j - 1];
          dist[j - 1] = sw ? dist[j] : dist[j - 1];
          id[j - 1]   = sw ? id[j]   : id[j - 1];
          dist[j]     = sw ? fd : dist[j];
          id[j]       = sw ? fi : id[j];
        }
      }
    }
  }

  int* o = idx_out + ((size_t)(batch * N_ + q)) * K_;
#pragma unroll
  for (int j = 0; j < K_; ++j) o[j] = id[j];
}

// ---------------------------------------------------------------------------
// Kernel 2: pack W (1728x256 f32, row-major) into per-fragment f16 layout:
// Wp[chunk][ctile][lane][h], h-th half = W[chunk*32 + Kmap(h,lane/16)][ctile*16
// + lane%16], Kmap(h,g) = (h<8 ? h : h+8) + 8*g  (16-bit B-matrix VGPR layout).
// GEMM then reads each B fragment as one contiguous 32B load per lane.
// ---------------------------------------------------------------------------
__global__ __launch_bounds__(256) void pack_w_kernel(const float* __restrict__ W,
                                                     _Float16* __restrict__ Wp) {
  const int tid = blockIdx.x * 256 + threadIdx.x;          // 54*16*32 = 27648
  if (tid >= NCHUNK * 16 * 32) return;
  const int lane  = tid & 31;
  const int ct    = (tid >> 5) & 15;
  const int chunk = tid >> 9;
  const int group = lane >> 4;
  const int n     = ct * 16 + (lane & 15);

  _Float16* dst = Wp + (size_t)tid * 16;
#pragma unroll
  for (int h = 0; h < 16; ++h) {
    const int kloc = (h < 8 ? h : h + 8) + 8 * group;
    const int k    = chunk * 32 + kloc;
    dst[h] = (_Float16)W[(size_t)k * E_ + n];
  }
}

// ---------------------------------------------------------------------------
// Kernel 3: fused knn-gather + GEMM + bias, f16 WMMA with f32 accumulation.
// One wave per (16-row tile, 64-col group): 4 accumulators, A reused 4x.
// ---------------------------------------------------------------------------
__global__ __launch_bounds__(256) void gemm_kernel(const float* __restrict__ x,
                                                   const int* __restrict__ idx,
                                                   const _Float16* __restrict__ Wp,
                                                   const float* __restrict__ bias,
                                                   float* __restrict__ out) {
  const int lane     = threadIdx.x & 31;
  const int wid      = (blockIdx.x << 3) + (threadIdx.x >> 5); // 0..8191
  const int row_tile = wid >> 2;                               // 0..2047
  const int colgrp   = wid & 3;                                // 64 cols each
  const int group    = lane >> 4;
  const int M        = lane & 15;

  const int row   = row_tile * 16 + M;                         // 0..32767
  const int batch = row >> 13;                                 // /8192
  const float* xb = x + (size_t)batch * N_ * D_;
  const int* nbr  = idx + (size_t)row * K_;

  v8f acc[4] = {};

  for (int chunk = 0; chunk < NCHUNK; ++chunk) {
    const int kk    = chunk >> 1;                 // neighbor 0..26
    const int dbase = ((chunk & 1) << 5) + (group << 3);
    const int nb    = nbr[kk];
    const float* src = xb + (size_t)nb * D_ + dbase;

    // A fragment: lane holds row M, halves h0..7 -> K = 8g..8g+7 (contig),
    // h8..15 -> K = 16+8g..16+8g+7 (contig). Two 8-float runs per lane.
    const f32x4 f0 = *(const f32x4*)(src);
    const f32x4 f1 = *(const f32x4*)(src + 4);
    const f32x4 f2 = *(const f32x4*)(src + 16);
    const f32x4 f3 = *(const f32x4*)(src + 20);

    v16h a;
#pragma unroll
    for (int e = 0; e < 4; ++e) {
      a[e]      = (_Float16)f0[e];
      a[4 + e]  = (_Float16)f1[e];
      a[8 + e]  = (_Float16)f2[e];
      a[12 + e] = (_Float16)f3[e];
    }

    const _Float16* wbase =
        Wp + (((size_t)chunk * 16 + colgrp * 4) * 32 + lane) * 16;
#pragma unroll
    for (int j = 0; j < 4; ++j) {
      const v16h bfrag = *(const v16h*)(wbase + (size_t)j * 32 * 16);
      acc[j] = __builtin_amdgcn_wmma_f32_16x16x32_f16(
          /*neg_a=*/false, a, /*neg_b=*/false, bfrag,
          /*c_mod=*/(short)0, acc[j], /*reuse_a=*/false, /*reuse_b=*/false);
    }
  }

  // Epilogue: C/D layout -> lane holds col N=lane%16, rows r + 8*(lane/16).
  const int colb = colgrp * 64;
#pragma unroll
  for (int j = 0; j < 4; ++j) {
    const int col = colb + j * 16 + M;
    const float bv = bias[col];
#pragma unroll
    for (int r = 0; r < 8; ++r) {
      const int orow = row_tile * 16 + r + 8 * group;
      out[(size_t)orow * E_ + col] = acc[j][r] + bv;
    }
  }
}

// ---------------------------------------------------------------------------
extern "C" void kernel_launch(void* const* d_in, const int* in_sizes, int n_in,
                              void* d_out, int out_size, void* d_ws, size_t ws_size,
                              hipStream_t stream) {
  const float* x    = (const float*)d_in[0];   // (B,N,64)  f32
  const float* xv   = (const float*)d_in[1];   // (B,N,3)   f32
  const float* W    = (const float*)d_in[2];   // (1728,256) f32
  const float* bias = (const float*)d_in[3];   // (256,)     f32
  float* out        = (float*)d_out;           // (B,N,256)  f32

  int* idx      = (int*)d_ws;                              // 3,538,944 B
  _Float16* Wp  = (_Float16*)((char*)d_ws + (4u << 20));   //   884,736 B

  knn_kernel <<<128,  256, 0, stream>>>(xv, idx);
  pack_w_kernel<<<108, 256, 0, stream>>>(W, Wp);
  gemm_kernel<<<1024, 256, 0, stream>>>(x, idx, Wp, bias, out);
}
